// BlockGlobalSelfAttention_87256555586202
// MI455X (gfx1250) — matrix operand order repaired and checked by reading, not compiled, not run on hardware
//
#include <hip/hip_runtime.h>
#include <hip/hip_bf16.h>
#include <math.h>

// ---------------- problem constants ----------------
constexpr int N_   = 2;
constexpr int T_   = 4096;
constexpr int HID_ = 1024;
constexpr int NH_  = 16;
constexpr int D_   = 64;
constexpr float NEG_ = -10000.0f;
constexpr float NEGINF_ = -1.0e30f;

typedef __attribute__((ext_vector_type(16))) _Float16 v16h;
typedef __attribute__((ext_vector_type(8)))  float    v8f;

// ---------------- WMMA helpers (gfx1250, wave32) ----------------
__device__ __forceinline__ v8f wmma16(v16h a, v16h b, v8f c) {
  // v_wmma_f32_16x16x32_f16
  return __builtin_amdgcn_wmma_f32_16x16x32_f16(false, a, false, b, (short)0, c,
                                                false, false);
}

// ---------------- CDNA5 async global->LDS copy (ASYNCcnt path) ----------------
// GLOBAL_LOAD_ASYNC_TO_LDS_B128: copies global memory into LDS, 16B/lane,
// tracked with ASYNCcnt (CDNA5 ISA). Builtin not
// probe-verified, so use inline asm. Low 32 bits of a generic LDS pointer are
// the LDS byte address (aperture truncation rule, CDNA5 ISA).
__device__ __forceinline__ void async_ld_b128(void* lds_dst, const void* gsrc) {
  unsigned l = (unsigned)(uintptr_t)lds_dst;
  asm volatile("global_load_async_to_lds_b128 %0, %1, off"
               :
               : "v"(l), "v"(gsrc)
               : "memory");
}
__device__ __forceinline__ void wait_asynccnt0() {
  asm volatile("s_wait_asynccnt 0" ::: "memory");
}

// A fragment: 16x32 f16, row-major source with leading dim `ld` (halves).
// Layout used here (contiguous k, same as B loaders): lane m=l&15 holds row m; lanes 0-15 K=0..15, 16-31 K=16..31.
__device__ __forceinline__ v16h load_frag_a(const _Float16* src, int ld, int row, int col) {
  int l = threadIdx.x & 31, m = l & 15, hi = l >> 4;
  v16h a;
#pragma unroll
  for (int j = 0; j < 8; ++j) {
    int k = hi * 16 + 2 * j;
    a[2 * j]     = src[(size_t)(row + m) * ld + col + k];
    a[2 * j + 1] = src[(size_t)(row + m) * ld + col + k + 1];
  }
  return a;
}

// B fragment 32x16 where B[k][n] = src[(kBase+k)*ld + nBase+n] (e.g. V: keys x d).
// lanes 0-15 hold K=0..15, lanes 16-31 hold K=16..31, 2 packed per VGPR.
__device__ __forceinline__ v16h load_frag_b(const _Float16* src, int ld, int kBase, int nBase) {
  int l = threadIdx.x & 31, n = l & 15, hi = l >> 4;
  v16h b;
#pragma unroll
  for (int j = 0; j < 8; ++j) {
    int k = hi * 16 + 2 * j;
    b[2 * j]     = src[(size_t)(kBase + k) * ld + nBase + n];
    b[2 * j + 1] = src[(size_t)(kBase + k + 1) * ld + nBase + n];
  }
  return b;
}

// B fragment 32x16 built transposed: B[k][n] = src[(nBase+n)*ld + kBase+k]
// (e.g. K^T for Q@K^T, or W^T for X@W^T), src row-major (rows x ld).
__device__ __forceinline__ v16h load_frag_b_trans(const _Float16* src, int ld, int nBase, int kBase) {
  int l = threadIdx.x & 31, n = l & 15, hi = l >> 4;
  v16h b;
#pragma unroll
  for (int j = 0; j < 8; ++j) {
    int k = hi * 16 + 2 * j;
    b[2 * j]     = src[(size_t)(nBase + n) * ld + kBase + k];
    b[2 * j + 1] = src[(size_t)(nBase + n) * ld + kBase + k + 1];
  }
  return b;
}

__device__ __forceinline__ float lane16_max(float v) {
#pragma unroll
  for (int m = 1; m < 16; m <<= 1) v = fmaxf(v, __shfl_xor(v, m, 32));
  return v;
}
__device__ __forceinline__ float lane16_sum(float v) {
#pragma unroll
  for (int m = 1; m < 16; m <<= 1) v += __shfl_xor(v, m, 32);
  return v;
}

// ---------------- 1) QKV projection: out = X @ W^T + b (f16 out, (n,h,t,d)) ----
// Register-double-buffered: global loads for tile k+1 issue right after the
// barrier and retire behind the 4 WMMAs of tile k (one s_wait_loadcnt per
// iteration instead of one per 16B chunk).
constexpr int XLD = 40;  // padded LDS leading dims (avoid bank conflicts)
constexpr int WLD = 40;

__global__ __launch_bounds__(256, 1) void qkv_gemm(const float* __restrict__ X,
                                                   const float* __restrict__ W,
                                                   const float* __restrict__ bias,
                                                   _Float16* __restrict__ out) {
  __shared__ _Float16 Xs[128 * XLD];
  __shared__ _Float16 Ws[64 * WLD];
  int tid = threadIdx.x;
  int w = tid >> 5, l = tid & 31, col = l & 15, rh = l >> 4;
  int blockRow = blockIdx.x * 128, colBase = blockIdx.y * 64;

  float4 xreg[4], wreg[2];
  auto load_tiles = [&](int kk) {
#pragma unroll
    for (int it = 0; it < 4; ++it) {
      int lin = tid + 256 * it, r = lin >> 3, c4 = (lin & 7) << 2;
      xreg[it] = *(const float4*)(X + (size_t)(blockRow + r) * HID_ + kk + c4);
    }
#pragma unroll
    for (int it = 0; it < 2; ++it) {
      int lin = tid + 256 * it, r = lin >> 3, c4 = (lin & 7) << 2;
      wreg[it] = *(const float4*)(W + (size_t)(colBase + r) * HID_ + kk + c4);
    }
  };
  auto store_tiles = [&]() {
#pragma unroll
    for (int it = 0; it < 4; ++it) {
      int lin = tid + 256 * it, r = lin >> 3, c4 = (lin & 7) << 2;
      _Float16* dst = Xs + r * XLD + c4;
      dst[0] = (_Float16)xreg[it].x; dst[1] = (_Float16)xreg[it].y;
      dst[2] = (_Float16)xreg[it].z; dst[3] = (_Float16)xreg[it].w;
    }
#pragma unroll
    for (int it = 0; it < 2; ++it) {
      int lin = tid + 256 * it, r = lin >> 3, c4 = (lin & 7) << 2;
      _Float16* dst = Ws + r * WLD + c4;
      dst[0] = (_Float16)wreg[it].x; dst[1] = (_Float16)wreg[it].y;
      dst[2] = (_Float16)wreg[it].z; dst[3] = (_Float16)wreg[it].w;
    }
  };

  v8f acc[4] = {};
  load_tiles(0);
  for (int kk = 0; kk < HID_; kk += 32) {
    store_tiles();          // f32->f16 convert + LDS
    __syncthreads();
    if (kk + 32 < HID_) load_tiles(kk + 32);  // overlap next tile with WMMAs
    v16h a = load_frag_a(Xs, XLD, w * 16, 0);
#pragma unroll
    for (int nt = 0; nt < 4; ++nt)
      acc[nt] = wmma16(a, load_frag_b_trans(Ws, WLD, nt * 16, 0), acc[nt]);
    __syncthreads();        // protect LDS from next iteration's stores
  }
#pragma unroll
  for (int nt = 0; nt < 4; ++nt) {
    int o = colBase + nt * 16 + col;
    float bo = bias[o];
    int h = o >> 6, d = o & 63;
#pragma unroll
    for (int j = 0; j < 8; ++j) {
      int m = blockRow + w * 16 + rh * 8 + j;
      int n = m >> 12, t = m & 4095;
      out[(((size_t)(n * NH_ + h)) * T_ + t) * D_ + d] = (_Float16)(acc[nt][j] + bo);
    }
  }
}

// ---------------- 2) norm-based top-k selection per 64-token block ------------
__global__ __launch_bounds__(64, 1) void select_topk(const _Float16* __restrict__ kh,
                                                     const float* __restrict__ bq,
                                                     const float* __restrict__ am,
                                                     int* __restrict__ gidx,
                                                     int* __restrict__ lidx) {
  int b = blockIdx.x;  // n*NH*64 + h*64 + blk
  int blk = b & 63, nh = b >> 6, h = nh & 15, n = nh >> 4;
  int i = threadIdx.x;  // 0..63
  int tok = blk * 64 + i;
  const _Float16* kr = kh + ((size_t)nh * T_ + tok) * D_;
  float s = 0.f;
#pragma unroll
  for (int d = 0; d < D_; ++d) {
    float x = (float)kr[d] + bq[h * D_ + d];
    s += x * x;
  }
  if (am[n * T_ + tok] != 0.f) s = 0.f;  // norms *= valid
  __shared__ float norms[64];
  __shared__ int flags[64];
  norms[i] = s;
  __syncthreads();
  // stable ascending rank (matches jnp.argsort tie-breaking)
  int pos = 0;
  for (int j = 0; j < 64; ++j) {
    float nj = norms[j];
    pos += (nj < s) || (nj == s && j < i);
  }
  int top = (pos >= 48) ? 1 : 0;
  flags[i] = top;
  __syncthreads();
  int cnt = 0;
  for (int j = 0; j < i; ++j) cnt += top ? flags[j] : (1 - flags[j]);
  if (top) gidx[(size_t)nh * 1024 + blk * 16 + cnt] = tok;
  else     lidx[(size_t)nh * 3072 + blk * 48 + cnt] = tok;
}

// ---------------- 3) block-local attention pass (templated) -------------------
// CHUNK: query rows per workgroup; WIN: gathered-window keys; CK: keys per block;
// KPAD: staged/padded key count (mult of 32); GKEY: append k[:, :, 0] with mask 0.
template <int CHUNK, int WIN, int CK, int KPAD, bool GKEY>
__global__ __launch_bounds__(256, 1) void attn_pass(const _Float16* __restrict__ qh,
                                                    const _Float16* __restrict__ kh,
                                                    const _Float16* __restrict__ vh,
                                                    const int* __restrict__ idx, int idxLen,
                                                    const float* __restrict__ am,
                                                    float* __restrict__ ctxOut,
                                                    float* __restrict__ lseOut) {
  constexpr int NT = KPAD / 16;   // 16-wide key tiles
  constexpr int KCH = KPAD / 32;  // 32-deep PV chunks
  extern __shared__ char smem[];
  _Float16* Ks = (_Float16*)smem;                  // [KPAD][64]
  _Float16* Vs = Ks + (size_t)KPAD * 64;           // [KPAD][64]
  float*    Msk = (float*)(Vs + (size_t)KPAD * 64);// [KPAD]
  _Float16* Pw = (_Float16*)(Msk + KPAD);          // [CHUNK/16 waves][16][KPAD]

  int tid = threadIdx.x;
  int w = tid >> 5, l = tid & 31, col = l & 15, rh = l >> 4;
  int qb = blockIdx.x, h = blockIdx.y, n = blockIdx.z;
  int nh = n * NH_ + h;
  size_t nhRow = (size_t)nh * T_;

  // ---- stage gathered K/V rows + additive mask into LDS ----
  // Real rows go through the CDNA5 async global->LDS path (no VGPR bounce);
  // window/alignment padding rows are zero-filled with plain DS stores.
  uint4 z4 = make_uint4(0u, 0u, 0u, 0u);
  for (int j = tid; j < KPAD; j += blockDim.x) {
    int tok = -1;
    float mval = NEGINF_;  // pure alignment padding -> -inf-ish
    if (j < WIN) {
      int jj = (qb - 1) * CK + j;
      if (jj >= 0 && jj < idxLen) {
        tok = idx[(size_t)nh * idxLen + jj];
        mval = am[n * T_ + tok];
      } else {
        mval = NEG_;  // reference window padding: zero key, -10000 mask
      }
    } else if (GKEY && j == WIN) {
      tok = 0;       // appended global (BOS) key, mask 0
      mval = 0.f;
    }
    Msk[j] = mval;
    _Float16* kd = Ks + (size_t)j * 64;
    _Float16* vd = Vs + (size_t)j * 64;
    if (tok >= 0) {
      const char* ksrc = (const char*)(kh + (nhRow + tok) * D_);
      const char* vsrc = (const char*)(vh + (nhRow + tok) * D_);
#pragma unroll
      for (int c = 0; c < 8; ++c) {
        async_ld_b128((char*)kd + c * 16, ksrc + c * 16);
        async_ld_b128((char*)vd + c * 16, vsrc + c * 16);
      }
    } else {
#pragma unroll
      for (int c = 0; c < 8; ++c) {
        ((uint4*)kd)[c] = z4;
        ((uint4*)vd)[c] = z4;
      }
    }
  }
  wait_asynccnt0();   // drain this wave's async copies; barrier covers DS
  __syncthreads();

  // ---- scores: S = Q(16x64) @ K^T, kept entirely in VGPRs (wave32, big RF) ----
  int rowBase = qb * CHUNK + w * 16;
  const _Float16* qptr = qh + nhRow * D_;
  v16h aq0 = load_frag_a(qptr, D_, rowBase, 0);
  v16h aq1 = load_frag_a(qptr, D_, rowBase, 32);
  v8f S[NT];
#pragma unroll
  for (int kt = 0; kt < NT; ++kt) {
    v8f c = {};
    c = wmma16(aq0, load_frag_b_trans(Ks, D_, kt * 16, 0), c);
    c = wmma16(aq1, load_frag_b_trans(Ks, D_, kt * 16, 32), c);
    float mk = Msk[kt * 16 + col];
#pragma unroll
    for (int j = 0; j < 8; ++j) c[j] = c[j] * 0.125f + mk;  // 1/sqrt(64)
    S[kt] = c;
  }

  // ---- row softmax: max/sum via 16-lane shuffles; P -> LDS as f16 ----
  _Float16* PwW = Pw + (size_t)w * 16 * KPAD;
  float sums[8];
#pragma unroll
  for (int j = 0; j < 8; ++j) {
    float mx = -3.0e38f;
#pragma unroll
    for (int kt = 0; kt < NT; ++kt) mx = fmaxf(mx, S[kt][j]);
    mx = lane16_max(mx);
    float sm = 0.f;
#pragma unroll
    for (int kt = 0; kt < NT; ++kt) {
      float p = __expf(S[kt][j] - mx);
      sm += p;
      PwW[(size_t)(rh * 8 + j) * KPAD + kt * 16 + col] = (_Float16)p;
    }
    sm = lane16_sum(sm);
    sums[j] = sm;
    if (col == 0)  // lanes 0 and 16 cover all 16 rows
      lseOut[nhRow + rowBase + rh * 8 + j] = mx + __logf(sm);
  }
  __syncthreads();

  // ---- ctx = P(16xKPAD) @ V(KPADx64) ----
  v8f ctx[4] = {};
#pragma unroll
  for (int kc = 0; kc < KCH; ++kc) {
    v16h ap = load_frag_a(PwW, KPAD, 0, kc * 32);
#pragma unroll
    for (int dt = 0; dt < 4; ++dt)
      ctx[dt] = wmma16(ap, load_frag_b(Vs, D_, kc * 32, dt * 16), ctx[dt]);
  }
#pragma unroll
  for (int dt = 0; dt < 4; ++dt) {
    int d = dt * 16 + col;
#pragma unroll
    for (int j = 0; j < 8; ++j) {
      int t = rowBase + rh * 8 + j;
      ctxOut[(nhRow + t) * D_ + d] = ctx[dt][j] / sums[j];
    }
  }
}

// ---------------- 4) BOS token: full attention, unscaled scores ---------------
__global__ __launch_bounds__(256, 1) void bos_attn(const _Float16* __restrict__ qh,
                                                   const _Float16* __restrict__ kh,
                                                   const _Float16* __restrict__ vh,
                                                   const float* __restrict__ am,
                                                   float* __restrict__ bos) {
  int nh = blockIdx.x, n = nh >> 4;
  int tid = threadIdx.x, w = tid >> 5, l = tid & 31;
  __shared__ float q0s[64];
  __shared__ float wmax[8], wsum[8], wacc[8][64], gstat[2];
  if (tid < 64) q0s[tid] = (float)qh[((size_t)nh * T_) * D_ + tid];
  __syncthreads();
  float sc[16], mx = -3.0e38f;
  for (int i = 0; i < 16; ++i) {
    int t = tid + (i << 8);
    const _Float16* kr = kh + ((size_t)nh * T_ + t) * D_;
    float s = 0.f;
#pragma unroll
    for (int d = 0; d < D_; ++d) s += q0s[d] * (float)kr[d];
    s += am[n * T_ + t];  // no 1/sqrt(d) for the BOS pass (per reference)
    sc[i] = s;
    mx = fmaxf(mx, s);
  }
#pragma unroll
  for (int m = 16; m >= 1; m >>= 1) mx = fmaxf(mx, __shfl_xor(mx, m, 32));
  if (l == 0) wmax[w] = mx;
  __syncthreads();
  if (tid == 0) {
    float m2 = wmax[0];
    for (int i = 1; i < 8; ++i) m2 = fmaxf(m2, wmax[i]);
    gstat[0] = m2;
  }
  __syncthreads();
  mx = gstat[0];
  float acc[64];
#pragma unroll
  for (int d = 0; d < D_; ++d) acc[d] = 0.f;
  float sm = 0.f;
  for (int i = 0; i < 16; ++i) {
    int t = tid + (i << 8);
    float p = __expf(sc[i] - mx);
    sm += p;
    const _Float16* vr = vh + ((size_t)nh * T_ + t) * D_;
#pragma unroll
    for (int d = 0; d < D_; ++d) acc[d] += p * (float)vr[d];
  }
#pragma unroll
  for (int m = 16; m >= 1; m >>= 1) sm += __shfl_xor(sm, m, 32);
#pragma unroll
  for (int d = 0; d < D_; ++d)
#pragma unroll
    for (int m = 16; m >= 1; m >>= 1) acc[d] += __shfl_xor(acc[d], m, 32);
  if (l == 0) {
    wsum[w] = sm;
    for (int d = 0; d < D_; ++d) wacc[w][d] = acc[d];
  }
  __syncthreads();
  if (tid == 0) {
    float s2 = 0.f;
    for (int i = 0; i < 8; ++i) s2 += wsum[i];
    gstat[1] = s2;
  }
  __syncthreads();
  if (tid < 64) {
    float c = 0.f;
    for (int i = 0; i < 8; ++i) c += wacc[i][tid];
    bos[(size_t)nh * D_ + tid] = c / gstat[1];
  }
}

// ---------------- 5) merge: sigmoid(lse_l - lse_g) blend + BOS row ------------
__global__ __launch_bounds__(256, 1) void merge_out(const float* __restrict__ ctx_l,
                                                    const float* __restrict__ ctx_g,
                                                    const float* __restrict__ lse_l,
                                                    const float* __restrict__ lse_g,
                                                    const float* __restrict__ bos,
                                                    float* __restrict__ out) {
  size_t i = (size_t)blockIdx.x * blockDim.x + threadIdx.x;
  if (i >= (size_t)N_ * T_ * HID_) return;
  int o = (int)(i & (HID_ - 1));
  size_t nt = i >> 10;
  int t = (int)(nt & (T_ - 1)), n = (int)(nt >> 12);
  int h = o >> 6, d = o & 63;
  float val;
  if (t == 0) {
    val = bos[(size_t)(n * NH_ + h) * D_ + d];
  } else {
    size_t row = (size_t)(n * NH_ + h) * T_ + t;
    float cg = ctx_g[row * D_ + d], cl = ctx_l[row * D_ + d];
    float p = 1.f / (1.f + __expf(lse_g[row] - lse_l[row]));
    val = cg + p * (cl - cg);
  }
  out[i] = val;
}

// ---------------- host launcher ----------------
extern "C" void kernel_launch(void* const* d_in, const int* in_sizes, int n_in,
                              void* d_out, int out_size, void* d_ws, size_t ws_size,
                              hipStream_t stream) {
  (void)in_sizes; (void)n_in; (void)out_size; (void)ws_size;
  const float* hs = (const float*)d_in[0];
  const float* am = (const float*)d_in[1];
  const float* Wq = (const float*)d_in[2];
  const float* bq = (const float*)d_in[3];
  const float* Wk = (const float*)d_in[4];
  const float* bk = (const float*)d_in[5];
  const float* Wv = (const float*)d_in[6];
  const float* bv = (const float*)d_in[7];
  float* out = (float*)d_out;

  // workspace layout (~113.5 MiB total)
  char* p = (char*)d_ws;
  auto alloc = [&](size_t bytes) {
    char* r = p;
    p += (bytes + 255) & ~(size_t)255;
    return r;
  };
  constexpr size_t QKV_ELEMS = (size_t)N_ * NH_ * T_ * D_;  // 8,388,608
  _Float16* qhp = (_Float16*)alloc(QKV_ELEMS * 2);
  _Float16* khp = (_Float16*)alloc(QKV_ELEMS * 2);
  _Float16* vhp = (_Float16*)alloc(QKV_ELEMS * 2);
  int* gidxp = (int*)alloc((size_t)N_ * NH_ * 1024 * 4);
  int* lidxp = (int*)alloc((size_t)N_ * NH_ * 3072 * 4);
  float* ctxl = (float*)alloc(QKV_ELEMS * 4);
  float* ctxg = (float*)alloc(QKV_ELEMS * 4);
  float* lsel = (float*)alloc((size_t)N_ * NH_ * T_ * 4);
  float* lseg = (float*)alloc((size_t)N_ * NH_ * T_ * 4);
  float* bosp = (float*)alloc((size_t)N_ * NH_ * D_ * 4);

  // 1) projections: M=8192 rows, 1024 cols each
  dim3 gg(64, 16);
  qkv_gemm<<<gg, 256, 0, stream>>>(hs, Wq, bq, qhp);
  qkv_gemm<<<gg, 256, 0, stream>>>(hs, Wk, bk, khp);
  qkv_gemm<<<gg, 256, 0, stream>>>(hs, Wv, bv, vhp);

  // 2) top-k selection: one block per (n,h,64-token chunk)
  select_topk<<<N_ * NH_ * (T_ / 64), 64, 0, stream>>>(khp, bq, am, gidxp, lidxp);

  // 3) BOS row
  bos_attn<<<N_ * NH_, 256, 0, stream>>>(qhp, khp, vhp, am, bosp);

  // 4) local pass: chunk 128, window 3*96=288 keys + global key, padded to 320
  constexpr size_t shmem_local =
      (size_t)320 * 64 * 2 * 2 + 320 * 4 + (size_t)128 * 320 * 2;  // 165,120 B
  attn_pass<128, 288, 96, 320, true>
      <<<dim3(T_ / 128, NH_, N_), 256, shmem_local, stream>>>(
          qhp, khp, vhp, lidxp, 3072, am, ctxl, lsel);

  // 5) global pass: chunk 64, window 3*16=48 keys, padded to 64
  constexpr size_t shmem_global =
      (size_t)64 * 64 * 2 * 2 + 64 * 4 + (size_t)64 * 64 * 2;  // 24,832 B
  attn_pass<64, 48, 16, 64, false>
      <<<dim3(T_ / 64, NH_, N_), 128, shmem_global, stream>>>(
          qhp, khp, vhp, gidxp, 1024, am, ctxg, lseg);

  // 6) merge + BOS overwrite
  size_t total = (size_t)N_ * T_ * HID_;
  merge_out<<<(unsigned)((total + 255) / 256), 256, 0, stream>>>(
      ctxl, ctxg, lsel, lseg, bosp, out);
}
